// MultiHeadAttention_28114855920396
// MI455X (gfx1250) — compile-verified, matrix operand order checked by reading, not run on hardware
//
#include <hip/hip_runtime.h>
#include <hip/hip_bf16.h>
#include <math.h>

// ---------------------------------------------------------------------------
// Types for CDNA5 WMMA (wave32): 16x16x32 bf16 -> f32 accumulate
// ---------------------------------------------------------------------------
typedef __attribute__((ext_vector_type(16))) __bf16 v16bf;
typedef __attribute__((ext_vector_type(8)))  __bf16 v8bf;
typedef __attribute__((ext_vector_type(8)))  float  v8f;

constexpr int    BATCH      = 2;
constexpr int    S          = 8192;   // n * heads (flattened sequence)
constexpr int    Dh         = 64;     // head dim
constexpr int    E          = 512;    // embed dim
constexpr int    F          = 1024;   // MLP hidden
constexpr int    NROW       = 2048;   // BATCH * 1024 token rows for MLP
constexpr float  ATT_SCALE  = 0.125f; // 1/sqrt(64)
constexpr size_t ATTN_ELEMS = (size_t)BATCH * S * S;

// ---------------------------------------------------------------------------
// Fragment loaders. ISA 16-bit A/B 16x32 layout:
//   lane L: row/col = L%16, half = L/16
//   element i (of 16 bf16): K = (i/8)*16 + half*8 + (i%8)
// => two contiguous 8-element (16B) chunks per lane at k0+half*8 and k0+16+half*8
// ---------------------------------------------------------------------------
__device__ __forceinline__ v16bf load_tile_bf16(const __bf16* __restrict__ base,
                                                int ld, int row0, int k0) {
  const int lane = threadIdx.x & 31;
  const int r    = lane & 15;
  const int half = lane >> 4;
  const __bf16* p = base + (size_t)(row0 + r) * ld + k0 + half * 8;
  union { v16bf v; v8bf h[2]; } u;
  u.h[0] = *(const v8bf*)(p);
  u.h[1] = *(const v8bf*)(p + 16);
  return u.v;
}

__device__ __forceinline__ v16bf load_tile_f32_bf16(const float* __restrict__ base,
                                                    int ld, int row0, int k0) {
  const int lane = threadIdx.x & 31;
  const int r    = lane & 15;
  const int half = lane >> 4;
  const float* p = base + (size_t)(row0 + r) * ld + k0 + half * 8;
  v8f c0 = *(const v8f*)(p);
  v8f c1 = *(const v8f*)(p + 16);
  v16bf out;
#pragma unroll
  for (int i = 0; i < 8; ++i) { out[i] = (__bf16)c0[i]; out[8 + i] = (__bf16)c1[i]; }
  return out;
}

__device__ __forceinline__ v8f wmma_bf16(v16bf a, v16bf b, v8f c) {
  return __builtin_amdgcn_wmma_f32_16x16x32_bf16(false, a, false, b, (short)0, c,
                                                 false, false);
}

#define V8F_ZERO {0.f,0.f,0.f,0.f,0.f,0.f,0.f,0.f}

// ---------------------------------------------------------------------------
// Kernel 0: f32 -> bf16 conversion
// ---------------------------------------------------------------------------
__global__ void cvt_f32_bf16_kernel(const float* __restrict__ in,
                                    __bf16* __restrict__ out, int n) {
  int i = blockIdx.x * blockDim.x + threadIdx.x;
  if (i < n) out[i] = (__bf16)in[i];
}

// ---------------------------------------------------------------------------
// Kernel 1: Q/K/V projections.  Q = X @ Wq^T etc.  X:[b,S,64] bf16.
// One wave -> one 16(s) x 16(e) tile of Q, K and V (K sweep 64 = 2 WMMA each).
// V is written TRANSPOSED: Vt[b][e][t] so attn@V B-loads are contiguous.
// ---------------------------------------------------------------------------
__global__ void __launch_bounds__(128)
qkv_proj_kernel(const __bf16* __restrict__ xh, const __bf16* __restrict__ wq,
                const __bf16* __restrict__ wk, const __bf16* __restrict__ wv,
                __bf16* __restrict__ Q, __bf16* __restrict__ K,
                __bf16* __restrict__ Vt) {
  const int gwid = blockIdx.x * 4 + (threadIdx.x >> 5);
  const int n0 = (gwid & 3) * 16;
  const int s0 = ((gwid >> 2) & 511) * 16;
  const int b  = gwid >> 11;
  const __bf16* X = xh + (size_t)b * S * Dh;

  v8f aq = V8F_ZERO, ak = V8F_ZERO, av = V8F_ZERO;
#pragma unroll
  for (int k0 = 0; k0 < Dh; k0 += 32) {
    v16bf a = load_tile_bf16(X, Dh, s0, k0);
    aq = wmma_bf16(a, load_tile_bf16(wq, Dh, n0, k0), aq);
    ak = wmma_bf16(a, load_tile_bf16(wk, Dh, n0, k0), ak);
    av = wmma_bf16(a, load_tile_bf16(wv, Dh, n0, k0), av);
  }
  const int lane = threadIdx.x & 31;
  const int n = lane & 15, half = lane >> 4;
  __bf16* Qb = Q  + (size_t)b * S * Dh;
  __bf16* Kb = K  + (size_t)b * S * Dh;
  __bf16* Vb = Vt + (size_t)b * Dh * S;
#pragma unroll
  for (int r = 0; r < 8; ++r) {
    const int m = s0 + r + 8 * half;
    Qb[(size_t)m * Dh + n0 + n] = (__bf16)aq[r];
    Kb[(size_t)m * Dh + n0 + n] = (__bf16)ak[r];
    Vb[(size_t)(n0 + n) * S + m] = (__bf16)av[r];
  }
}

// ---------------------------------------------------------------------------
// Kernel 2: scores = (Q K^T) * scale, f32, written straight into d_out attn.
// Block = 8 waves sharing ONE 128-row K strip (16 KB), staged into LDS with
// GLOBAL_LOAD_ASYNC_TO_LDS_B128 (ASYNCcnt), drained with s_wait_asynccnt.
// Each wave then computes a 16(s) x 128(t) strip, B fragments from LDS.
// ---------------------------------------------------------------------------
__global__ void __launch_bounds__(256)
scores_kernel(const __bf16* __restrict__ Q, const __bf16* __restrict__ K,
              float* __restrict__ attn) {
  __shared__ __bf16 kbuf[128 * Dh];     // K[t0:t0+128, 0:64] bf16 = 16 KB
  const int blk  = blockIdx.x;          // 0..8191
  const int t0   = (blk & 63) * 128;
  const int sblk = ((blk >> 6) & 63) * 128;
  const int b    = blk >> 12;
  const __bf16* Qb = Q + (size_t)b * S * Dh;
  const __bf16* Kb = K + (size_t)b * S * Dh;
  float* A = attn + (size_t)b * S * S;

  // Cooperative async stage: 1024 x 16B chunks, 4 per thread.
  {
    const __bf16* src = Kb + (size_t)t0 * Dh;
    const int c = threadIdx.x * 4;
#pragma unroll
    for (int i = 0; i < 4; ++i) {
      const unsigned lds = (unsigned)(size_t)(&kbuf[(c + i) * 8]);
      const __bf16* g = src + (size_t)(c + i) * 8;
      asm volatile("global_load_async_to_lds_b128 %0, %1, off"
                   :: "v"(lds), "v"(g) : "memory");
    }
    asm volatile("s_wait_asynccnt 0x0" ::: "memory");
    __syncthreads();
  }

  const int wid = threadIdx.x >> 5;
  const int s0  = sblk + wid * 16;
  const v16bf a0 = load_tile_bf16(Qb, Dh, s0, 0);
  const v16bf a1 = load_tile_bf16(Qb, Dh, s0, 32);
  const int lane = threadIdx.x & 31;
  const int n = lane & 15, half = lane >> 4;

  for (int j = 0; j < 8; ++j) {
    const int tl = j * 16;                       // row inside LDS strip
    v8f acc = V8F_ZERO;
    acc = wmma_bf16(a0, load_tile_bf16(kbuf, Dh, tl, 0),  acc);
    acc = wmma_bf16(a1, load_tile_bf16(kbuf, Dh, tl, 32), acc);
    const int tt = t0 + tl;
#pragma unroll
    for (int r = 0; r < 8; ++r)
      A[(size_t)(s0 + r + 8 * half) * S + tt + n] = acc[r] * ATT_SCALE;
  }
}

// ---------------------------------------------------------------------------
// Kernel 3: row softmax over 8192-wide rows, in place in d_out.
// 256 threads/row, 32 values/thread in registers, LDS tree reductions.
// ---------------------------------------------------------------------------
__global__ void __launch_bounds__(256)
softmax_kernel(float* __restrict__ attn) {
  __shared__ float red[256];
  float* p = attn + (size_t)blockIdx.x * S;
  const int t = threadIdx.x;

  float vals[32];
  float m = -3.402823466e38f;
#pragma unroll
  for (int i = 0; i < 32; ++i) { vals[i] = p[t + i * 256]; m = fmaxf(m, vals[i]); }

  red[t] = m; __syncthreads();
  for (int s2 = 128; s2 > 0; s2 >>= 1) {
    if (t < s2) red[t] = fmaxf(red[t], red[t + s2]);
    __syncthreads();
  }
  m = red[0]; __syncthreads();

  float sum = 0.f;
#pragma unroll
  for (int i = 0; i < 32; ++i) { vals[i] = expf(vals[i] - m); sum += vals[i]; }

  red[t] = sum; __syncthreads();
  for (int s2 = 128; s2 > 0; s2 >>= 1) {
    if (t < s2) red[t] += red[t + s2];
    __syncthreads();
  }
  const float inv = 1.0f / red[0];
#pragma unroll
  for (int i = 0; i < 32; ++i) p[t + i * 256] = vals[i] * inv;
}

// ---------------------------------------------------------------------------
// Kernel 4: out = attn @ V.  One wave -> 16(s) x 64(d) output: 4 accumulators,
// so each f32 attn fragment is loaded ONCE (512 MB total, not 2 GB) and feeds
// 4 WMMA.  B fragments come from transposed Vt (contiguous, L2-resident).
// ---------------------------------------------------------------------------
__global__ void __launch_bounds__(128)
attn_v_kernel(const float* __restrict__ attn, const __bf16* __restrict__ Vt,
              __bf16* __restrict__ ao) {
  const int gwid = blockIdx.x * 4 + (threadIdx.x >> 5);
  const int s0 = (gwid & 511) * 16;
  const int b  = gwid >> 9;
  const float*  A  = attn + (size_t)b * S * S;
  const __bf16* Vb = Vt   + (size_t)b * Dh * S;
  const int lane = threadIdx.x & 31;
  const int n = lane & 15, half = lane >> 4;

  v8f acc0 = V8F_ZERO, acc1 = V8F_ZERO, acc2 = V8F_ZERO, acc3 = V8F_ZERO;
  for (int k0 = 0; k0 < S; k0 += 32) {
    if (k0 + 64 < S)  // uniform; global_prefetch for the A rows two tiles ahead
      __builtin_prefetch(A + (size_t)(s0 + n) * S + k0 + 64, 0, 0);
    v16bf a = load_tile_f32_bf16(A, S, s0, k0);
    acc0 = wmma_bf16(a, load_tile_bf16(Vb, S, 0,  k0), acc0);
    acc1 = wmma_bf16(a, load_tile_bf16(Vb, S, 16, k0), acc1);
    acc2 = wmma_bf16(a, load_tile_bf16(Vb, S, 32, k0), acc2);
    acc3 = wmma_bf16(a, load_tile_bf16(Vb, S, 48, k0), acc3);
  }

  v8f accs[4] = {acc0, acc1, acc2, acc3};
#pragma unroll
  for (int nt = 0; nt < 4; ++nt) {
#pragma unroll
    for (int r = 0; r < 8; ++r) {
      const int s   = s0 + r + 8 * half;
      const int row = s >> 3;                      // token index
      const int col = (s & 7) * Dh + nt * 16 + n;  // head*64 + d
      ao[((size_t)b * 1024 + row) * E + col] = (__bf16)accs[nt][r];
    }
  }
}

// ---------------------------------------------------------------------------
// Kernel 5: h = GELU_exact(ao @ W1^T + b1), bf16 out.  K sweep 512 -> 16 WMMA.
// ---------------------------------------------------------------------------
__global__ void __launch_bounds__(128)
mlp1_kernel(const __bf16* __restrict__ ao, const __bf16* __restrict__ w1,
            const float* __restrict__ b1, __bf16* __restrict__ h) {
  const int gwid = blockIdx.x * 4 + (threadIdx.x >> 5);
  const int f0 = (gwid & 63) * 16;
  const int m0 = (gwid >> 6) * 16;

  v8f acc = V8F_ZERO;
  for (int k0 = 0; k0 < E; k0 += 32)
    acc = wmma_bf16(load_tile_bf16(ao, E, m0, k0),
                    load_tile_bf16(w1, E, f0, k0), acc);

  const int lane = threadIdx.x & 31;
  const int n = lane & 15, half = lane >> 4;
  const float bias = b1[f0 + n];
#pragma unroll
  for (int r = 0; r < 8; ++r) {
    const float v = acc[r] + bias;
    const float g = 0.5f * v * (1.0f + erff(v * 0.70710678118654752f));
    h[(size_t)(m0 + r + 8 * half) * F + f0 + n] = (__bf16)g;
  }
}

// ---------------------------------------------------------------------------
// Kernel 6: out = h @ W2^T + b2, f32 to d_out.  K sweep 1024 -> 32 WMMA.
// ---------------------------------------------------------------------------
__global__ void __launch_bounds__(128)
mlp2_kernel(const __bf16* __restrict__ h, const __bf16* __restrict__ w2,
            const float* __restrict__ b2, float* __restrict__ out) {
  const int gwid = blockIdx.x * 4 + (threadIdx.x >> 5);
  const int e0 = (gwid & 31) * 16;
  const int m0 = (gwid >> 5) * 16;

  v8f acc = V8F_ZERO;
  for (int k0 = 0; k0 < F; k0 += 32)
    acc = wmma_bf16(load_tile_bf16(h, F, m0, k0),
                    load_tile_bf16(w2, F, e0, k0), acc);

  const int lane = threadIdx.x & 31;
  const int n = lane & 15, half = lane >> 4;
  const float bias = b2[e0 + n];
#pragma unroll
  for (int r = 0; r < 8; ++r)
    out[(size_t)(m0 + r + 8 * half) * E + e0 + n] = acc[r] + bias;
}

// ---------------------------------------------------------------------------
// Launch
// ---------------------------------------------------------------------------
extern "C" void kernel_launch(void* const* d_in, const int* in_sizes, int n_in,
                              void* d_out, int out_size, void* d_ws, size_t ws_size,
                              hipStream_t stream) {
  const float* x  = (const float*)d_in[0];
  const float* wq = (const float*)d_in[1];
  const float* wk = (const float*)d_in[2];
  const float* wv = (const float*)d_in[3];
  const float* w1 = (const float*)d_in[4];
  const float* b1 = (const float*)d_in[5];
  const float* w2 = (const float*)d_in[6];
  const float* b2 = (const float*)d_in[7];

  float* attn = (float*)d_out;                 // [2, 8192, 8192]
  float* out  = (float*)d_out + ATTN_ELEMS;    // [2, 1024, 512]

  // Workspace layout (bf16, ~17 MB total; every region 16B aligned)
  __bf16* p   = (__bf16*)d_ws;
  __bf16* xh  = p; p += (size_t)BATCH * S * Dh;  // x reinterpreted [2,8192,64]
  __bf16* wqb = p; p += Dh * Dh;
  __bf16* wkb = p; p += Dh * Dh;
  __bf16* wvb = p; p += Dh * Dh;
  __bf16* Qb  = p; p += (size_t)BATCH * S * Dh;
  __bf16* Kb  = p; p += (size_t)BATCH * S * Dh;
  __bf16* Vt  = p; p += (size_t)BATCH * Dh * S;  // V transposed [b][64][8192]
  __bf16* w1b = p; p += (size_t)F * E;
  __bf16* w2b = p; p += (size_t)E * F;
  __bf16* ao  = p; p += (size_t)NROW * E;        // attn output [2,1024,512]
  __bf16* hb  = p; p += (size_t)NROW * F;        // MLP hidden  [2,1024,1024]

  auto cvt = [&](const float* src, __bf16* dst, int n) {
    cvt_f32_bf16_kernel<<<(n + 255) / 256, 256, 0, stream>>>(src, dst, n);
  };
  cvt(x,  xh,  BATCH * S * Dh);   // reshape b n (h d) -> b (n h) d is flat-identical
  cvt(wq, wqb, Dh * Dh);
  cvt(wk, wkb, Dh * Dh);
  cvt(wv, wvb, Dh * Dh);
  cvt(w1, w1b, F * E);
  cvt(w2, w2b, E * F);

  qkv_proj_kernel<<<1024, 128, 0, stream>>>(xh, wqb, wkb, wvb, Qb, Kb, Vt);
  scores_kernel  <<<8192, 256, 0, stream>>>(Qb, Kb, attn);
  softmax_kernel <<<BATCH * S, 256, 0, stream>>>(attn);
  attn_v_kernel  <<<256, 128, 0, stream>>>(attn, Vt, ao);
  mlp1_kernel    <<<2048, 128, 0, stream>>>(ao, w1b, b1, hb);
  mlp2_kernel    <<<1024, 128, 0, stream>>>(hb, w2b, b2, out);
}
